// GraphTransformerLayer_78700980731990
// MI455X (gfx1250) — compile-verified, matrix operand order checked by reading, not run on hardware
//
#include <hip/hip_runtime.h>
#include <hip/hip_bf16.h>

#define NN   4096
#define DD   256
#define NH   4
#define HDIM 64
#define LN_EPS 1e-5f
#define NEGBIG (-3.0e38f)

typedef __attribute__((ext_vector_type(16))) _Float16 v16h;
typedef __attribute__((ext_vector_type(8)))  float    v8f;

// ---------------------------------------------------------------------------
// WMMA helpers (CDNA5 gfx1250, wave32): D = A(16x32 f16) * B(32x16 f16) + C(f32)
// ---------------------------------------------------------------------------
__device__ __forceinline__ v8f wmma_step(v16h a, v16h b, v8f c) {
  return __builtin_amdgcn_wmma_f32_16x16x32_f16(
      /*neg_a=*/false, a, /*neg_b=*/false, b,
      /*c_mod=*/(short)0, c, /*reuse_a=*/false, /*reuse_b=*/false);
}

// A-fragment loader (also used for B via the transposed operand):
// lane L (L<16) holds row M=L, K = kb+0..7 in elems 0..7 and kb+16..23 in 8..15,
// lane L+16 holds K = kb+8..15 / kb+24..31 (kb = (L>=16)*8). Each half is a
// contiguous 16-byte run -> vectorizable global_load.
__device__ __forceinline__ v16h frag_load(const _Float16* __restrict__ base,
                                          int ld, int row0, int k0) {
  const int lane = threadIdx.x & 31;
  const int r    = lane & 15;
  const int kb   = (lane >> 4) << 3;
  const _Float16* p = base + (size_t)(row0 + r) * (size_t)ld + (size_t)(k0 + kb);
  v16h f;
#pragma unroll
  for (int e = 0; e < 8; ++e)  f[e] = p[e];
#pragma unroll
  for (int e = 8; e < 16; ++e) f[e] = p[e + 8];
  return f;
}

// ---------------------------------------------------------------------------
// Small conversion kernels
// ---------------------------------------------------------------------------
__global__ void cvt_f16_kernel(const float* __restrict__ in,
                               _Float16* __restrict__ out, int n) {
  int i = blockIdx.x * blockDim.x + threadIdx.x;
  if (i < n) out[i] = (_Float16)in[i];
}

// W is K x N row-major; WT becomes N x K row-major (f16)
__global__ void transpose_w_kernel(const float* __restrict__ W,
                                   _Float16* __restrict__ WT, int K, int Nc) {
  int i = blockIdx.x * blockDim.x + threadIdx.x;
  if (i < K * Nc) {
    int k = i / Nc, n = i - k * Nc;
    WT[(size_t)n * K + k] = (_Float16)W[i];
  }
}

// ---------------------------------------------------------------------------
// Generic WMMA GEMM: C[M,Nc] = act(A[M,K] * BT[Nc,K]^T + bias) (+ residual)
// One 16x16 output tile per wave, 8 waves per block, K consumed 32/step.
// ---------------------------------------------------------------------------
__global__ void gemm16_kernel(const _Float16* __restrict__ A,
                              const _Float16* __restrict__ BT,
                              const float* __restrict__ bias,
                              const float* __restrict__ residual,
                              float* __restrict__ Cf,
                              _Float16* __restrict__ Ch,
                              _Float16* __restrict__ CT,
                              int M, int Nc, int K, int relu) {
  const int wave  = threadIdx.x >> 5;
  const int lane  = threadIdx.x & 31;
  const int tile  = blockIdx.x * 8 + wave;
  const int ntCol = Nc >> 4;
  const int tm    = tile / ntCol;
  const int tn    = tile - tm * ntCol;
  const int m0    = tm << 4;
  const int n0    = tn << 4;
  if (m0 >= M) return;

  v8f c = {};
  for (int k = 0; k < K; k += 32) {
    v16h a = frag_load(A,  K, m0, k);
    v16h b = frag_load(BT, K, n0, k);
    c = wmma_step(a, b, c);
  }

  const int col    = n0 + (lane & 15);
  const int rowoff = (lane >> 4) << 3;
  const float bn   = bias ? bias[col] : 0.0f;
#pragma unroll
  for (int r = 0; r < 8; ++r) {
    int row = m0 + r + rowoff;
    float v = c[r] + bn;
    if (relu) v = fmaxf(v, 0.0f);
    if (residual) v += residual[(size_t)row * Nc + col];
    if (Cf) Cf[(size_t)row * Nc + col] = v;
    if (Ch) Ch[(size_t)row * Nc + col] = (_Float16)v;
    if (CT) CT[(size_t)col * M + row]  = (_Float16)v;   // transposed store (for V)
  }
}

// ---------------------------------------------------------------------------
// Flash-style masked multi-head attention.
// Block = one 16-row i-tile, 4 waves = 4 heads (adj rows shared via cache).
// Per 32-wide j block: S = Q_i K_j^T (2x2 WMMA), mask by adj, online softmax,
// P staged through LDS (C-layout -> A-layout), O += P V (4 WMMA on Vt).
// ---------------------------------------------------------------------------
__global__ void attn_kernel(const _Float16* __restrict__ Qh,
                            const _Float16* __restrict__ Kh,
                            const _Float16* __restrict__ Vt,   // [DD][NN] f16
                            const int* __restrict__ adj,
                            _Float16* __restrict__ Oh) {
  __shared__ _Float16 plds[NH][16 * 32];

  const int h    = threadIdx.x >> 5;      // head = wave id
  const int lane = threadIdx.x & 31;
  const int i0   = blockIdx.x << 4;
  const int colr   = lane & 15;
  const int rowoff = (lane >> 4) << 3;
  const int hc   = h * HDIM;

  // Q fragments for this i-tile / head (K dim = 64 -> two 16x32 fragments)
  v16h qa0 = frag_load(Qh, DD, i0, hc);
  v16h qa1 = frag_load(Qh, DD, i0, hc + 32);

  v8f o0 = {}, o1 = {}, o2 = {}, o3 = {};
  float rm[8], rs[8];
#pragma unroll
  for (int r = 0; r < 8; ++r) { rm[r] = NEGBIG; rs[r] = 0.0f; }

  _Float16* lp = plds[h];

  for (int jb = 0; jb < NN; jb += 32) {
    // ---- scores: two 16x16 j-subtiles
    v8f s0 = {}, s1 = {};
    s0 = wmma_step(qa0, frag_load(Kh, DD, jb,      hc),      s0);
    s0 = wmma_step(qa1, frag_load(Kh, DD, jb,      hc + 32), s0);
    s1 = wmma_step(qa0, frag_load(Kh, DD, jb + 16, hc),      s1);
    s1 = wmma_step(qa1, frag_load(Kh, DD, jb + 16, hc + 32), s1);

    // ---- scale + adjacency mask
#pragma unroll
    for (int r = 0; r < 8; ++r) {
      const int* arow = adj + (size_t)(i0 + r + rowoff) * NN + jb;
      float a0 = s0[r] * 0.125f;           // 1/sqrt(64)
      float a1 = s1[r] * 0.125f;
      if (arow[colr]      == 0) a0 = NEGBIG;
      if (arow[16 + colr] == 0) a1 = NEGBIG;
      s0[r] = a0; s1[r] = a1;
    }

    // ---- online softmax update (row = r + rowoff; 16 cols live in a 16-lane group)
#pragma unroll
    for (int r = 0; r < 8; ++r) {
      float mx = fmaxf(s0[r], s1[r]);
#pragma unroll
      for (int off = 1; off < 16; off <<= 1)
        mx = fmaxf(mx, __shfl_xor(mx, off, 32));
      float mnew = fmaxf(rm[r], mx);
      float fac  = __expf(rm[r] - mnew);
      float p0 = __expf(s0[r] - mnew);
      float p1 = __expf(s1[r] - mnew);
      s0[r] = p0; s1[r] = p1;
      float bs = p0 + p1;
#pragma unroll
      for (int off = 1; off < 16; off <<= 1)
        bs += __shfl_xor(bs, off, 32);
      rs[r] = rs[r] * fac + bs;
      rm[r] = mnew;
      o0[r] *= fac; o1[r] *= fac; o2[r] *= fac; o3[r] *= fac;
    }

    // ---- stage P (16x32) through LDS: C-layout -> A-layout
#pragma unroll
    for (int r = 0; r < 8; ++r) {
      int row = r + rowoff;
      lp[row * 32 + colr]      = (_Float16)s0[r];
      lp[row * 32 + 16 + colr] = (_Float16)s1[r];
    }
    asm volatile("s_wait_dscnt 0" ::: "memory");
    v16h pa = frag_load(lp, 32, 0, 0);

    // ---- O += P * V  (Vt is [d][j] so B-fragments stream contiguously over j)
    o0 = wmma_step(pa, frag_load(Vt, NN, hc,      jb), o0);
    o1 = wmma_step(pa, frag_load(Vt, NN, hc + 16, jb), o1);
    o2 = wmma_step(pa, frag_load(Vt, NN, hc + 32, jb), o2);
    o3 = wmma_step(pa, frag_load(Vt, NN, hc + 48, jb), o3);
  }

  // ---- normalize and emit f16 attention output (consumed by Wo GEMM)
#pragma unroll
  for (int r = 0; r < 8; ++r) {
    int row = i0 + r + rowoff;
    float inv = 1.0f / rs[r];
    _Float16* dst = Oh + (size_t)row * DD + hc + colr;
    dst[0]  = (_Float16)(o0[r] * inv);
    dst[16] = (_Float16)(o1[r] * inv);
    dst[32] = (_Float16)(o2[r] * inv);
    dst[48] = (_Float16)(o3[r] * inv);
  }
}

// ---------------------------------------------------------------------------
// LayerNorm over D=256 (one block per row); residual already fused in GEMM.
// ---------------------------------------------------------------------------
__global__ void layernorm_kernel(const float* __restrict__ inp,
                                 const float* __restrict__ gamma,
                                 const float* __restrict__ beta,
                                 float* __restrict__ outF,
                                 _Float16* __restrict__ outH) {
  __shared__ float red[2][8];
  const int row = blockIdx.x;
  const int tid = threadIdx.x;
  float v = inp[(size_t)row * DD + tid];
  float s = v, sq = v * v;
#pragma unroll
  for (int off = 1; off < 32; off <<= 1) {
    s  += __shfl_xor(s,  off, 32);
    sq += __shfl_xor(sq, off, 32);
  }
  if ((tid & 31) == 0) { red[0][tid >> 5] = s; red[1][tid >> 5] = sq; }
  __syncthreads();
  float ts = 0.0f, tq = 0.0f;
#pragma unroll
  for (int w = 0; w < 8; ++w) { ts += red[0][w]; tq += red[1][w]; }
  float mu   = ts * (1.0f / DD);
  float var  = tq * (1.0f / DD) - mu * mu;
  float rstd = rsqrtf(var + LN_EPS);
  float y = (v - mu) * rstd * gamma[tid] + beta[tid];
  if (outF) outF[(size_t)row * DD + tid] = y;
  if (outH) outH[(size_t)row * DD + tid] = (_Float16)y;
}

// ---------------------------------------------------------------------------
// Host-side orchestration
// ---------------------------------------------------------------------------
extern "C" void kernel_launch(void* const* d_in, const int* in_sizes, int n_in,
                              void* d_out, int out_size, void* d_ws, size_t ws_size,
                              hipStream_t stream) {
  const float* x    = (const float*)d_in[0];
  const int*   adj  = (const int*)  d_in[1];
  const float* Wq   = (const float*)d_in[2];  const float* bq = (const float*)d_in[3];
  const float* Wk   = (const float*)d_in[4];  const float* bk = (const float*)d_in[5];
  const float* Wv   = (const float*)d_in[6];  const float* bv = (const float*)d_in[7];
  const float* Wo   = (const float*)d_in[8];  const float* bo = (const float*)d_in[9];
  const float* ln_g = (const float*)d_in[10]; const float* ln_b = (const float*)d_in[11];
  const float* W1   = (const float*)d_in[12]; const float* b1 = (const float*)d_in[13];
  const float* W2   = (const float*)d_in[14]; const float* b2 = (const float*)d_in[15];
  float* out = (float*)d_out;

  char* ws = (char*)d_ws;
  size_t off = 0;
  auto carve = [&](size_t bytes) -> void* {
    void* p = ws + off;
    off += (bytes + 255) & ~(size_t)255;
    return p;
  };

  const size_t ND  = (size_t)NN * DD;
  const size_t DDQ = (size_t)DD * DD;

  _Float16* xh    = (_Float16*)carve(ND  * 2);
  _Float16* WqT   = (_Float16*)carve(DDQ * 2);
  _Float16* WkT   = (_Float16*)carve(DDQ * 2);
  _Float16* WvT   = (_Float16*)carve(DDQ * 2);
  _Float16* WoT   = (_Float16*)carve(DDQ * 2);
  _Float16* W1T   = (_Float16*)carve(DDQ * 2);
  _Float16* W2T   = (_Float16*)carve(DDQ * 2);
  _Float16* Qh    = (_Float16*)carve(ND  * 2);
  _Float16* Kh    = (_Float16*)carve(ND  * 2);
  _Float16* Vt    = (_Float16*)carve(ND  * 2);   // [DD][NN] transposed
  _Float16* attnh = (_Float16*)carve(ND  * 2);
  float*    tmpF  = (float*)   carve(ND  * 4);   // pre-LN staging (reused)
  float*    hF    = (float*)   carve(ND  * 4);   // LN1 output (residual for LN2)
  _Float16* hH    = (_Float16*)carve(ND  * 2);
  _Float16* fh    = (_Float16*)carve(ND  * 2);
  (void)ws_size; (void)n_in; (void)in_sizes; (void)out_size;

  const int cvtBlocks = (int)(ND / 256);
  const int twBlocks  = (int)(DDQ / 256);
  const int gemmBlocks = (NN / 16) * (DD / 16) / 8;   // 8 waves/block

  // 0) precision/layout prep
  cvt_f16_kernel<<<cvtBlocks, 256, 0, stream>>>(x, xh, (int)ND);
  transpose_w_kernel<<<twBlocks, 256, 0, stream>>>(Wq, WqT, DD, DD);
  transpose_w_kernel<<<twBlocks, 256, 0, stream>>>(Wk, WkT, DD, DD);
  transpose_w_kernel<<<twBlocks, 256, 0, stream>>>(Wv, WvT, DD, DD);
  transpose_w_kernel<<<twBlocks, 256, 0, stream>>>(Wo, WoT, DD, DD);
  transpose_w_kernel<<<twBlocks, 256, 0, stream>>>(W1, W1T, DD, DD);
  transpose_w_kernel<<<twBlocks, 256, 0, stream>>>(W2, W2T, DD, DD);

  // 1) QKV projections (V stored transposed for the P*V WMMA stream)
  gemm16_kernel<<<gemmBlocks, 256, 0, stream>>>(xh, WqT, bq, nullptr,
      nullptr, Qh, nullptr, NN, DD, DD, 0);
  gemm16_kernel<<<gemmBlocks, 256, 0, stream>>>(xh, WkT, bk, nullptr,
      nullptr, Kh, nullptr, NN, DD, DD, 0);
  gemm16_kernel<<<gemmBlocks, 256, 0, stream>>>(xh, WvT, bv, nullptr,
      nullptr, nullptr, Vt, NN, DD, DD, 0);

  // 2) masked flash attention (block = i-tile, waves = heads)
  attn_kernel<<<NN / 16, 128, 0, stream>>>(Qh, Kh, Vt, adj, attnh);

  // 3) output projection + residual(x), then LayerNorm -> h
  gemm16_kernel<<<gemmBlocks, 256, 0, stream>>>(attnh, WoT, bo, x,
      tmpF, nullptr, nullptr, NN, DD, DD, 0);
  layernorm_kernel<<<NN, DD, 0, stream>>>(tmpF, ln_g, ln_b, hF, hH);

  // 4) FFN: relu(h W1 + b1) W2 + b2 + h, then final LayerNorm -> out
  gemm16_kernel<<<gemmBlocks, 256, 0, stream>>>(hH, W1T, b1, nullptr,
      nullptr, fh, nullptr, NN, DD, DD, 1);
  gemm16_kernel<<<gemmBlocks, 256, 0, stream>>>(fh, W2T, b2, hF,
      tmpF, nullptr, nullptr, NN, DD, DD, 0);
  layernorm_kernel<<<NN, DD, 0, stream>>>(tmpF, ln_g, ln_b, out, nullptr);
}
